// LocalModel_39256001085596
// MI455X (gfx1250) — matrix operand order repaired and checked by reading, not compiled, not run on hardware
//
#include <hip/hip_runtime.h>

// ---------------------------------------------------------------------------
// MI455X (gfx1250, wave32) implementation of the 6-layer shared-weight local-
// attention encoder. Matrix math in bf16 WMMA (v_wmma_f32_16x16x32_bf16) with
// f32 accumulation; attention rewritten as banded attention + V-sum using the
// identity  softmax(where(mask,s,0)) @ V
//   = (sum_{inW}(e-1) v + sum_all v) / (sum_{inW}(e-1) + S).
// K-band staged into LDS via gfx1250 async-to-LDS copies (ASYNCcnt).
// ---------------------------------------------------------------------------

typedef __bf16 bf16_t;
typedef bf16_t bf16x16 __attribute__((ext_vector_type(16)));
typedef bf16_t bf16x8  __attribute__((ext_vector_type(8)));
typedef float  f32x8   __attribute__((ext_vector_type(8)));

union Frag {
  bf16x16 v;
  bf16x8  h[2];
};

__device__ __forceinline__ f32x8 zero8() {
  f32x8 z = {0.f, 0.f, 0.f, 0.f, 0.f, 0.f, 0.f, 0.f};
  return z;
}

// ------------------------- f32 -> bf16 conversion --------------------------
__global__ void f2bf_kernel(const float* __restrict__ in, bf16_t* __restrict__ out, int n) {
  int i = blockIdx.x * 256 + threadIdx.x;
  if (i < n) out[i] = (bf16_t)in[i];
}

// ------------------- embedding gather + positional enc ---------------------
__global__ __launch_bounds__(256)
void embed_kernel(const int* __restrict__ ids, const float* __restrict__ emb,
                  float* __restrict__ xF, bf16_t* __restrict__ xH) {
  const int row = blockIdx.x;           // b*S + s
  const int s   = row & 1023;
  const int p   = threadIdx.x;          // 0..255 pairs
  const int tok = ids[row];
  const float div = __expf(-9.210340371976184f * (float)(2 * p) * (1.0f / 512.0f));
  const float ang = (float)s * div;
  const float v0 = emb[(size_t)tok * 512 + 2 * p]     + __sinf(ang);
  const float v1 = emb[(size_t)tok * 512 + 2 * p + 1] + __cosf(ang);
  const size_t o = (size_t)row * 512 + 2 * p;
  xF[o] = v0;  xF[o + 1] = v1;
  xH[o] = (bf16_t)v0;  xH[o + 1] = (bf16_t)v1;
}

// ------------------------------- layernorm ---------------------------------
__global__ __launch_bounds__(256)
void ln_kernel(const float* __restrict__ in, const float* __restrict__ g,
               const float* __restrict__ beta, float* __restrict__ outF,
               bf16_t* __restrict__ outH) {
  __shared__ float s1[256], s2[256];
  const int row = blockIdx.x;
  const int tid = threadIdx.x;
  const float* x = in + (size_t)row * 512;
  const float x0 = x[tid], x1 = x[tid + 256];
  s1[tid] = x0 + x1;
  s2[tid] = x0 * x0 + x1 * x1;
  __syncthreads();
  for (int off = 128; off > 0; off >>= 1) {
    if (tid < off) { s1[tid] += s1[tid + off]; s2[tid] += s2[tid + off]; }
    __syncthreads();
  }
  const float mean = s1[0] * (1.0f / 512.0f);
  const float var  = s2[0] * (1.0f / 512.0f) - mean * mean;
  const float inv  = rsqrtf(var + 1e-5f);
  const float y0 = (x0 - mean) * inv * g[tid]       + beta[tid];
  const float y1 = (x1 - mean) * inv * g[tid + 256] + beta[tid + 256];
  outF[(size_t)row * 512 + tid]       = y0;
  outF[(size_t)row * 512 + tid + 256] = y1;
  outH[(size_t)row * 512 + tid]       = (bf16_t)y0;
  outH[(size_t)row * 512 + tid + 256] = (bf16_t)y1;
}

// -------------------------------- WMMA GEMM --------------------------------
// C[M,N] = A[M,K](bf16) @ B[N,K]^T(bf16) + bias (+ resid) (relu?)
// One wave per block; 32x64 output tile (2 M-tiles x 4 N-tiles, 8 accums).
// Each B fragment feeds 2 WMMAs; each A fragment feeds 4.
// Fragment layouts per CDNA5 ISA 7.12.2:
//   A (16x32 bf16): lane m=lr; K chunks at half*8 and 16+half*8
//   B (32x16 bf16): lane n=lr; K chunks at half*8 and 16+half*8
//   C/D f32: lane n=lr; VGPR v <-> m = v + 8*half
__global__ __launch_bounds__(32)
void gemm_bf16(const bf16_t* __restrict__ A, const bf16_t* __restrict__ Bm,
               const float* __restrict__ bias, const float* __restrict__ resid,
               float* __restrict__ outF, bf16_t* __restrict__ outH,
               int M, int N, int K, int relu) {
  const int lane = threadIdx.x & 31;
  const int half = lane >> 4;
  const int lr   = lane & 15;
  const int n0 = blockIdx.x * 64;
  const int m0 = blockIdx.y * 32;

  f32x8 acc[2][4];
#pragma unroll
  for (int r = 0; r < 2; ++r)
#pragma unroll
    for (int t = 0; t < 4; ++t) acc[r][t] = zero8();

  const bf16_t* ap0 = A  + (size_t)(m0 + lr) * K + half * 8;
  const bf16_t* ap1 = ap0 + (size_t)16 * K;
  const bf16_t* bp0 = Bm + (size_t)(n0 + lr) * K + half * 8;

  for (int k0 = 0; k0 < K; k0 += 32) {
    Frag a0, a1;
    a0.h[0] = *(const bf16x8*)(ap0 + k0);
    a0.h[1] = *(const bf16x8*)(ap0 + k0 + 16);
    a1.h[0] = *(const bf16x8*)(ap1 + k0);
    a1.h[1] = *(const bf16x8*)(ap1 + k0 + 16);
#pragma unroll
    for (int t = 0; t < 4; ++t) {
      Frag bb;
      const bf16_t* bp = bp0 + (size_t)t * 16 * K + k0;
      bb.h[0] = *(const bf16x8*)(bp);
      bb.h[1] = *(const bf16x8*)(bp + 16);
      acc[0][t] = __builtin_amdgcn_wmma_f32_16x16x32_bf16(false, a0.v, false, bb.v,
                                                          (short)0, acc[0][t], false, false);
      acc[1][t] = __builtin_amdgcn_wmma_f32_16x16x32_bf16(false, a1.v, false, bb.v,
                                                          (short)0, acc[1][t], false, false);
    }
  }
#pragma unroll
  for (int r = 0; r < 2; ++r) {
#pragma unroll
    for (int t = 0; t < 4; ++t) {
      const int n = n0 + t * 16 + lr;
      const float bi = bias[n];
#pragma unroll
      for (int vv = 0; vv < 8; ++vv) {
        const int m = m0 + r * 16 + vv + 8 * half;
        float val = acc[r][t][vv] + bi;
        if (resid) val += resid[(size_t)m * N + n];
        if (relu)  val = fmaxf(val, 0.0f);
        if (outF) outF[(size_t)m * N + n] = val;
        if (outH) outH[(size_t)m * N + n] = (bf16_t)val;
      }
    }
  }
}

// ----------------------- per-(b,head) V column sums ------------------------
__global__ __launch_bounds__(512)
void vsum_kernel(const bf16_t* __restrict__ v, float* __restrict__ vsum) {
  const int b = blockIdx.x;
  const int c = threadIdx.x;  // 0..511  (= h*64 + d)
  const bf16_t* vp = v + (size_t)b * 1024 * 512 + c;
  float acc = 0.f;
  for (int s = 0; s < 1024; ++s) acc += (float)vp[(size_t)s * 512];
  vsum[b * 512 + c] = acc;
}

// ----------------------------- banded attention ----------------------------
// grid = (S/64, B*H), 128 threads (4 waves). Wave w owns query rows 16w..16w+15.
// Stage 0: async-copy the 192x64 K band into LDS (GLOBAL_LOAD_ASYNC_TO_LDS_B128,
//          ASYNCcnt / s_wait_asynccnt).
// Phase 1: P = exp(scale * Q K^T) - 1 on the 64x192 band (WMMA, K from LDS),
//          rowsums via shfl_xor, P -> LDS (bf16).
// Stage 1: re-stage the same LDS buffer with V^T (64 d x 192 j).
// Phase 2: O = P @ V_band (WMMA, both operands from LDS), then
//          (O + Vsum) / (rowsum + S).
__global__ __launch_bounds__(128)
void attn_kernel(const bf16_t* __restrict__ q, const bf16_t* __restrict__ k,
                 const bf16_t* __restrict__ v, const float* __restrict__ vsum,
                 float* __restrict__ out) {
  __shared__ __align__(16) bf16_t KV[192 * 64];   // K band, then V^T band
  __shared__ __align__(16) bf16_t Pl[64 * 192];
  __shared__ float rs[64];

  const int b  = blockIdx.y >> 3;
  const int hh = blockIdx.y & 7;
  const int q0 = blockIdx.x * 64;
  const int jstart = q0 - 64;
  const int wave = threadIdx.x >> 5;
  const int lane = threadIdx.x & 31;
  const int half = lane >> 4;
  const int lr   = lane & 15;

  const size_t headoff = ((size_t)b * 1024) * 512 + hh * 64;

  // ---- stage 0: async K-band -> LDS (192 rows x 64 bf16 = 24 KB) ----
  // Low 32 bits of the flat shared-pointer == LDS byte offset (ISA 10.2).
  const unsigned ldsK = (unsigned)(size_t)(&KV[0]);
  {
#pragma unroll
    for (int i = 0; i < 12; ++i) {
      const int c  = i * 128 + (int)threadIdx.x;   // 16B chunk id, 1536 total
      const int jl = c >> 3;                       // band row (8 chunks/row)
      const int cc = c & 7;
      const int j  = jstart + jl;
      const int jc = j < 0 ? 0 : (j > 1023 ? 1023 : j);
      const bf16_t* gsrc = k + headoff + (size_t)jc * 512 + cc * 8;
      const unsigned ldsOff = ldsK + (unsigned)((jl * 64 + cc * 8) * 2);
      asm volatile("global_load_async_to_lds_b128 %0, %1, off"
                   :: "v"(ldsOff), "v"((unsigned long long)(size_t)gsrc)
                   : "memory");
    }
  }

  // Q fragments (loop-invariant): rows q0+16w+lr, K chunks for k0=0 and k0=32.
  const bf16_t* qrow = q + headoff + (size_t)(q0 + wave * 16 + lr) * 512 + half * 8;
  Frag qa[2];
  qa[0].h[0] = *(const bf16x8*)(qrow);
  qa[0].h[1] = *(const bf16x8*)(qrow + 16);
  qa[1].h[0] = *(const bf16x8*)(qrow + 32);
  qa[1].h[1] = *(const bf16x8*)(qrow + 48);

  asm volatile("s_wait_asynccnt 0x0" ::: "memory");
  __syncthreads();

  // ---- phase 1: scores over the band ----
  float rsum[8];
#pragma unroll
  for (int i = 0; i < 8; ++i) rsum[i] = 0.f;

  for (int t = 0; t < 12; ++t) {
    f32x8 acc = zero8();
    const int j = jstart + t * 16 + lr;
    const bf16_t* krow = &KV[(t * 16 + lr) * 64 + half * 8];
#pragma unroll
    for (int k0 = 0; k0 < 2; ++k0) {
      Frag bb;
      bb.h[0] = *(const bf16x8*)(krow + k0 * 32);
      bb.h[1] = *(const bf16x8*)(krow + k0 * 32 + 16);
      acc = __builtin_amdgcn_wmma_f32_16x16x32_bf16(false, qa[k0].v, false, bb.v,
                                                    (short)0, acc, false, false);
    }
#pragma unroll
    for (int vv = 0; vv < 8; ++vv) {
      const int m = q0 + wave * 16 + vv + 8 * half;
      const bool inw = (j >= 0) && (j < 1024) && (j >= m - 64) && (j < m + 64);
      const float pv = inw ? (__expf(acc[vv] * 0.125f) - 1.0f) : 0.0f;
      rsum[vv] += pv;
      Pl[(wave * 16 + vv + 8 * half) * 192 + t * 16 + lr] = (bf16_t)pv;
    }
  }
#pragma unroll
  for (int vv = 0; vv < 8; ++vv) {
    float r = rsum[vv];
    r += __shfl_xor(r, 1, 32);
    r += __shfl_xor(r, 2, 32);
    r += __shfl_xor(r, 4, 32);
    r += __shfl_xor(r, 8, 32);
    if (lr == 0) rs[wave * 16 + vv + 8 * half] = r;
  }
  __syncthreads();   // everyone done with K band + Pl/rs visible

  // ---- stage 1: V^T band into the same LDS buffer (64 d x 192 j) ----
  for (int idx = threadIdx.x; idx < 64 * 192; idx += 128) {
    const int d  = idx & 63;
    const int jl = idx >> 6;
    const int j  = jstart + jl;
    const int jc = j < 0 ? 0 : (j > 1023 ? 1023 : j);
    KV[d * 192 + jl] = v[headoff + (size_t)jc * 512 + d];
  }
  __syncthreads();

  // ---- phase 2: P @ V_band (both operands in LDS) ----
  f32x8 acc2[4];
#pragma unroll
  for (int t = 0; t < 4; ++t) acc2[t] = zero8();

  for (int kk = 0; kk < 192; kk += 32) {
    Frag a;
    const bf16_t* prow = &Pl[(wave * 16 + lr) * 192 + kk + half * 8];
    a.h[0] = *(const bf16x8*)(prow);
    a.h[1] = *(const bf16x8*)(prow + 16);
#pragma unroll
    for (int t = 0; t < 4; ++t) {
      Frag bb;
      const bf16_t* vp = &KV[(t * 16 + lr) * 192 + kk + half * 8];
      bb.h[0] = *(const bf16x8*)(vp);
      bb.h[1] = *(const bf16x8*)(vp + 16);
      acc2[t] = __builtin_amdgcn_wmma_f32_16x16x32_bf16(false, a.v, false, bb.v,
                                                        (short)0, acc2[t], false, false);
    }
  }
#pragma unroll
  for (int t = 0; t < 4; ++t) {
    const int d = t * 16 + lr;
    const float vs = vsum[b * 512 + hh * 64 + d];
#pragma unroll
    for (int vv = 0; vv < 8; ++vv) {
      const int ml = wave * 16 + vv + 8 * half;
      const float denom = rs[ml] + 1024.0f;
      out[headoff + (size_t)(q0 + ml) * 512 + d] = (acc2[t][vv] + vs) / denom;
    }
  }
}

// --------------------- classifier (two-stage, no atomics) ------------------
__global__ __launch_bounds__(256)
void cls1_kernel(const float* __restrict__ x, const float* __restrict__ w,
                 float* __restrict__ part) {
  const int bc = blockIdx.y;            // b*6 + c
  const int b = bc / 6, c = bc % 6;
  const int chunk = blockIdx.x;         // 64 chunks of 8192
  const float* xr = x + (size_t)b * 524288 + (size_t)chunk * 8192;
  const float* wr = w + (size_t)c * 524288 + (size_t)chunk * 8192;
  float a = 0.f;
  for (int i = threadIdx.x; i < 8192; i += 256) a += xr[i] * wr[i];
  __shared__ float sm[256];
  sm[threadIdx.x] = a;
  __syncthreads();
  for (int off = 128; off > 0; off >>= 1) {
    if (threadIdx.x < off) sm[threadIdx.x] += sm[threadIdx.x + off];
    __syncthreads();
  }
  if (threadIdx.x == 0) part[bc * 64 + chunk] = sm[0];
}

__global__ __launch_bounds__(64)
void cls2_kernel(const float* __restrict__ part, const float* __restrict__ bias,
                 float* __restrict__ out) {
  const int bc = blockIdx.x;
  const int c = bc % 6;
  __shared__ float sm[64];
  sm[threadIdx.x] = part[bc * 64 + threadIdx.x];
  __syncthreads();
  for (int off = 32; off > 0; off >>= 1) {
    if (threadIdx.x < off) sm[threadIdx.x] += sm[threadIdx.x + off];
    __syncthreads();
  }
  if (threadIdx.x == 0) out[bc] = sm[0] + bias[c];
}

// ---------------------------------------------------------------------------
extern "C" void kernel_launch(void* const* d_in, const int* in_sizes, int n_in,
                              void* d_out, int out_size, void* d_ws, size_t ws_size,
                              hipStream_t stream) {
  (void)in_sizes; (void)n_in; (void)out_size; (void)ws_size;

  const int B = 8, S = 1024, D = 512, HD = 2048;
  const int M = B * S;

  const int*   ids   = (const int*)  d_in[0];
  const float* emb   = (const float*)d_in[1];
  const float* wq    = (const float*)d_in[2];
  const float* bq    = (const float*)d_in[3];
  const float* wk    = (const float*)d_in[4];
  const float* bk    = (const float*)d_in[5];
  const float* wv    = (const float*)d_in[6];
  const float* bv    = (const float*)d_in[7];
  const float* ln_g  = (const float*)d_in[8];
  const float* ln_b  = (const float*)d_in[9];
  const float* fc1_w = (const float*)d_in[10];
  const float* fc1_b = (const float*)d_in[11];
  const float* fc2_w = (const float*)d_in[12];
  const float* fc2_b = (const float*)d_in[13];
  const float* out_w = (const float*)d_in[14];
  const float* out_b = (const float*)d_in[15];
  float* out = (float*)d_out;

  // workspace layout (~107 MB total)
  char* p = (char*)d_ws;
  auto alloc = [&](size_t bytes) -> char* {
    char* r = p;
    p += (bytes + 255) & ~(size_t)255;
    return r;
  };
  float*  xF   = (float*) alloc((size_t)M * D * 4);   // current x (f32)
  float*  tF   = (float*) alloc((size_t)M * D * 4);   // pre-LN temp (f32)
  bf16_t* xH   = (bf16_t*)alloc((size_t)M * D * 2);   // current x (bf16)
  bf16_t* qH   = (bf16_t*)alloc((size_t)M * D * 2);
  bf16_t* kH   = (bf16_t*)alloc((size_t)M * D * 2);
  bf16_t* vH   = (bf16_t*)alloc((size_t)M * D * 2);
  bf16_t* hH   = (bf16_t*)alloc((size_t)M * HD * 2);  // FFN hidden
  bf16_t* wqH  = (bf16_t*)alloc((size_t)D * D * 2);
  bf16_t* wkH  = (bf16_t*)alloc((size_t)D * D * 2);
  bf16_t* wvH  = (bf16_t*)alloc((size_t)D * D * 2);
  bf16_t* f1H  = (bf16_t*)alloc((size_t)HD * D * 2);
  bf16_t* f2H  = (bf16_t*)alloc((size_t)D * HD * 2);
  float*  vsum = (float*) alloc((size_t)B * D * 4);
  float*  part = (float*) alloc((size_t)48 * 64 * 4);

  // weight conversion to bf16 (once per launch; deterministic)
  f2bf_kernel<<<(D * D + 255) / 256, 256, 0, stream>>>(wq, wqH, D * D);
  f2bf_kernel<<<(D * D + 255) / 256, 256, 0, stream>>>(wk, wkH, D * D);
  f2bf_kernel<<<(D * D + 255) / 256, 256, 0, stream>>>(wv, wvH, D * D);
  f2bf_kernel<<<(HD * D + 255) / 256, 256, 0, stream>>>(fc1_w, f1H, HD * D);
  f2bf_kernel<<<(HD * D + 255) / 256, 256, 0, stream>>>(fc2_w, f2H, HD * D);

  embed_kernel<<<M, 256, 0, stream>>>(ids, emb, xF, xH);

  const dim3 gQKV(D / 64, M / 32);
  const dim3 gFC1(HD / 64, M / 32);
  const dim3 gFC2(D / 64, M / 32);
  for (int it = 0; it < 6; ++it) {
    gemm_bf16<<<gQKV, 32, 0, stream>>>(xH, wqH, bq, nullptr, nullptr, qH, M, D, D, 0);
    gemm_bf16<<<gQKV, 32, 0, stream>>>(xH, wkH, bk, nullptr, nullptr, kH, M, D, D, 0);
    gemm_bf16<<<gQKV, 32, 0, stream>>>(xH, wvH, bv, nullptr, nullptr, vH, M, D, D, 0);
    vsum_kernel<<<B, 512, 0, stream>>>(vH, vsum);
    attn_kernel<<<dim3(S / 64, B * 8), 128, 0, stream>>>(qH, kH, vH, vsum, tF);
    ln_kernel<<<M, 256, 0, stream>>>(tF, ln_g, ln_b, xF, xH);
    gemm_bf16<<<gFC1, 32, 0, stream>>>(xH, f1H, fc1_b, nullptr, nullptr, hH, M, HD, D, 1);
    gemm_bf16<<<gFC2, 32, 0, stream>>>(hH, f2H, fc2_b, xF, tF, nullptr, M, D, HD, 0);
    ln_kernel<<<M, 256, 0, stream>>>(tF, ln_g, ln_b, xF, xH);
  }

  cls1_kernel<<<dim3(64, 48), 256, 0, stream>>>(xF, out_w, part);
  cls2_kernel<<<48, 64, 0, stream>>>(part, out_b, out);
}
